// GCN_33346126086443
// MI455X (gfx1250) — matrix operand order repaired and checked by reading, not compiled, not run on hardware
//
#include <hip/hip_runtime.h>

// ---------------------------------------------------------------------------
// GCN (2-layer, PyG GCNConv semantics) for MI455X / gfx1250.
// Layer-1 GEMM (50000x128 @ 128x128) uses v_wmma_f32_16x16x32_bf16
// (bf16 inputs, f32 accumulate). Everything else is bandwidth/scatter bound.
// ---------------------------------------------------------------------------

typedef __attribute__((ext_vector_type(16))) __bf16 v16bf;
typedef __attribute__((ext_vector_type(8)))  __bf16 v8bf;
typedef __attribute__((ext_vector_type(8)))  float  v8f;

union AFrag { v16bf v; v8bf h[2]; };

__device__ __forceinline__ unsigned short f32_to_bf16(float f) {
    union { float f; unsigned u; } x; x.f = f;
    unsigned r = x.u + 0x7FFFu + ((x.u >> 16) & 1u);   // round-to-nearest-even
    return (unsigned short)(r >> 16);
}

// ---------------- degree / normalization --------------------------------
__global__ __launch_bounds__(256) void deg_init(float* deg, int N) {
    int i = blockIdx.x * blockDim.x + threadIdx.x;
    if (i < N) deg[i] = 1.0f;                  // self-loop
}

__global__ __launch_bounds__(256) void deg_accum(const int* __restrict__ dst,
                                                 float* deg, int E) {
    int e = blockIdx.x * blockDim.x + threadIdx.x;
    if (e < E) atomicAdd(&deg[dst[e]], 1.0f);
}

__global__ __launch_bounds__(256) void deg_finish(float* deg, int N) {
    int i = blockIdx.x * blockDim.x + threadIdx.x;
    if (i < N) deg[i] = rsqrtf(deg[i]);        // deg >= 1 always
}

// ---------------- layer-1 GEMM: lin1 = x @ W1 (bf16 WMMA, f32 acc) ------
#define LDS_S 136   // bf16 elements per row (272 B, 16-B aligned, pads banks)

__global__ __launch_bounds__(128) void gemm1_wmma(const float* __restrict__ x,
                                                  const float* __restrict__ W1,
                                                  float* __restrict__ lin1,
                                                  int N) {
    __shared__ unsigned short wt[128 * LDS_S];  // W1^T: wt[n][k] (bf16 bits)
    __shared__ unsigned short xs[64 * LDS_S];   // x tile  xs[r][k] (bf16 bits)

    const int t  = threadIdx.x;                 // 0..127
    const int r0 = blockIdx.x * 64;

    // Stage W1 transposed into LDS as bf16 (coalesced global reads).
    for (int k = 0; k < 128; ++k)
        wt[t * LDS_S + k] = f32_to_bf16(W1[k * 128 + t]);
    // Stage 64-row x tile as bf16 (coalesced per row, zero-fill OOB rows).
    for (int r = 0; r < 64; ++r) {
        int row = r0 + r;
        float v = (row < N) ? x[row * 128 + t] : 0.0f;
        xs[r * LDS_S + t] = f32_to_bf16(v);
    }
    __syncthreads();

    const int wave = t >> 5;            // 4 waves, 16 rows each
    const int lane = t & 31;
    const int m    = lane & 15;         // row (A) / col (B,C) within tile
    const int hi   = lane >> 4;         // lane-group half

    // A fragments for the 4 K-steps (16x32 bf16 each), per ISA lane layout:
    // half0 = K[s*32 + hi*8 .. +7], half1 = K[s*32 + hi*8 + 16 .. +7]
    AFrag a[4];
    const unsigned short* arow = &xs[(wave * 16 + m) * LDS_S];
#pragma unroll
    for (int s = 0; s < 4; ++s) {
        int kb = s * 32 + hi * 8;
        a[s].h[0] = *(const v8bf*)(const void*)(arow + kb);
        a[s].h[1] = *(const v8bf*)(const void*)(arow + kb + 16);
    }

    const bool full_tile = (r0 + 64 <= N);      // block-uniform branch

#pragma unroll
    for (int c = 0; c < 8; ++c) {       // 8 column tiles of 16
        const unsigned short* brow = &wt[(c * 16 + m) * LDS_S];
        v8f acc = {};
#pragma unroll
        for (int s = 0; s < 4; ++s) {
            AFrag b;
            int kb = s * 32 + hi * 8;   // B: lane holds K[kb .. kb+7], K[kb+16 .. kb+23] of col
            b.h[0] = *(const v8bf*)(const void*)(brow + kb);
            b.h[1] = *(const v8bf*)(const void*)(brow + kb + 16);
            acc = __builtin_amdgcn_wmma_f32_16x16x32_bf16(
                      false, a[s].v, false, b.v, (short)0, acc, false, false);
        }
        // C/D layout: VGPR j -> row (j + hi*8), column m
        const int rbase = r0 + wave * 16 + hi * 8;
        float* outp = lin1 + (size_t)rbase * 128 + c * 16 + m;
        if (full_tile) {                // uniform: straight-line stores
#pragma unroll
            for (int j = 0; j < 8; ++j)
                outp[(size_t)j * 128] = acc[j];
        } else {                        // only the final partial block
#pragma unroll
            for (int j = 0; j < 8; ++j)
                if (rbase + j < N) outp[(size_t)j * 128] = acc[j];
        }
    }
}

// ---------------- self-loop contribution: agg = lin1 * dinv^2 -----------
__global__ __launch_bounds__(256) void agg_init(const float* __restrict__ lin1,
                                                const float* __restrict__ dinv,
                                                float* __restrict__ agg, int N) {
    int t = blockIdx.x * blockDim.x + threadIdx.x;
    int node = t >> 5, lane = t & 31;
    if (node >= N) return;
    float di = dinv[node];
    float nrm = di * di;
    float4 v = ((const float4*)(lin1 + (size_t)node * 128))[lane];
    float4 r; r.x = v.x * nrm; r.y = v.y * nrm; r.z = v.z * nrm; r.w = v.w * nrm;
    ((float4*)(agg + (size_t)node * 128))[lane] = r;
}

// ---------------- edge scatter, layer 1 (one wave per edge) -------------
__global__ __launch_bounds__(256) void scatter1(const float* __restrict__ lin1,
                                                const float* __restrict__ dinv,
                                                const int* __restrict__ src,
                                                const int* __restrict__ dst,
                                                float* __restrict__ agg, int E) {
    int wid  = (blockIdx.x * blockDim.x + threadIdx.x) >> 5;
    int lane = threadIdx.x & 31;
    if (wid >= E) return;
    int s = src[wid], d = dst[wid];
    float nrm = dinv[s] * dinv[d];
    float4 v = ((const float4*)(lin1 + (size_t)s * 128))[lane];
    float* base = agg + (size_t)d * 128 + lane * 4;
    atomicAdd(base + 0, v.x * nrm);
    atomicAdd(base + 1, v.y * nrm);
    atomicAdd(base + 2, v.z * nrm);
    atomicAdd(base + 3, v.w * nrm);
}

// -------- per-node: h = relu(agg+b1); hv = h.W2; out = self-loop + b2 ---
__global__ __launch_bounds__(256) void node_reduce(const float* __restrict__ agg,
                                                   const float* __restrict__ b1,
                                                   const float* __restrict__ W2,
                                                   const float* __restrict__ dinv,
                                                   const float* __restrict__ b2,
                                                   float* __restrict__ hv,
                                                   float* __restrict__ out, int N) {
    int wid  = (blockIdx.x * blockDim.x + threadIdx.x) >> 5;
    int lane = threadIdx.x & 31;
    if (wid >= N) return;
    float4 av = ((const float4*)(agg + (size_t)wid * 128))[lane];
    float4 bv = ((const float4*)b1)[lane];
    float4 wv = ((const float4*)W2)[lane];
    float h0 = fmaxf(av.x + bv.x, 0.0f);
    float h1 = fmaxf(av.y + bv.y, 0.0f);
    float h2 = fmaxf(av.z + bv.z, 0.0f);
    float h3 = fmaxf(av.w + bv.w, 0.0f);
    float sum = h0 * wv.x + h1 * wv.y + h2 * wv.z + h3 * wv.w;
#pragma unroll
    for (int off = 16; off > 0; off >>= 1)
        sum += __shfl_xor(sum, off, 32);
    if (lane == 0) {
        hv[wid] = sum;
        float di = dinv[wid];
        out[wid] = sum * di * di + b2[0];     // self-loop term + bias
    }
}

// ---------------- edge scatter, layer 2 (scalar) ------------------------
__global__ __launch_bounds__(256) void scatter2(const int* __restrict__ src,
                                                const int* __restrict__ dst,
                                                const float* __restrict__ hv,
                                                const float* __restrict__ dinv,
                                                float* __restrict__ out, int E) {
    int e = blockIdx.x * blockDim.x + threadIdx.x;
    if (e < E) {
        int s = src[e], d = dst[e];
        atomicAdd(&out[d], hv[s] * dinv[s] * dinv[d]);
    }
}

// ---------------------------------------------------------------------------
extern "C" void kernel_launch(void* const* d_in, const int* in_sizes, int n_in,
                              void* d_out, int out_size, void* d_ws, size_t ws_size,
                              hipStream_t stream) {
    const float* x  = (const float*)d_in[0];
    const int*   ei = (const int*)d_in[1];      // [2, E]: src row then dst row
    const float* W1 = (const float*)d_in[2];
    const float* b1 = (const float*)d_in[3];
    const float* W2 = (const float*)d_in[4];
    const float* b2 = (const float*)d_in[5];
    float* out = (float*)d_out;

    const int N = in_sizes[0] / 128;
    const int E = in_sizes[1] / 2;
    const int* src = ei;
    const int* dst = ei + E;

    // Workspace layout (floats): dinv[N] | lin1[N*128] | agg[N*128] | hv[N]
    float* dinv = (float*)d_ws;
    float* lin1 = dinv + (size_t)N;
    float* agg  = lin1 + (size_t)N * 128;
    float* hv   = agg  + (size_t)N * 128;

    deg_init  <<<(N + 255) / 256, 256, 0, stream>>>(dinv, N);
    deg_accum <<<(E + 255) / 256, 256, 0, stream>>>(dst, dinv, E);
    deg_finish<<<(N + 255) / 256, 256, 0, stream>>>(dinv, N);

    gemm1_wmma<<<(N + 63) / 64, 128, 0, stream>>>(x, W1, lin1, N);

    agg_init  <<<(N * 32 + 255) / 256, 256, 0, stream>>>(lin1, dinv, agg, N);
    scatter1  <<<(E + 7) / 8,          256, 0, stream>>>(lin1, dinv, src, dst, agg, E);

    node_reduce<<<(N + 7) / 8, 256, 0, stream>>>(agg, b1, W2, dinv, b2, hv, out, N);
    scatter2   <<<(E + 255) / 256, 256, 0, stream>>>(src, dst, hv, dinv, out, E);
}
